// GCACostRegNetResample_80642305950161
// MI455X (gfx1250) — matrix operand order repaired and checked by reading, not compiled, not run on hardware
//
#include <hip/hip_runtime.h>

typedef __attribute__((ext_vector_type(16))) _Float16 v16h;
typedef __attribute__((ext_vector_type(8)))  _Float16 v8h;
typedef __attribute__((ext_vector_type(8)))  float    v8f;

#define Hh   128
#define Ww   160
#define Dd   32
#define Cc   8
#define K2   9
#define ICv  72      // valid input channels (C * K2)
#define ICp  96      // padded channel dim; layout kk-major: ic' = kk*8 + c
#define OCn  8
#define TW   4       // pixels (w) per block
#define NB   6       // halo columns (TW + 2)
#define HW   (Hh*Ww)
#define NCH  9       // 3 depth taps * 3 K-chunks of 32 (padded K = 288)

__device__ __forceinline__ int iclamp(int v, int lo, int hi) {
    return v < lo ? lo : (v > hi ? hi : v);
}

__global__ __launch_bounds__(256)
void gca_fused_kernel(const float* __restrict__ x,
                      const float* __restrict__ dp,
                      const float* __restrict__ normal,
                      const float* __restrict__ intri,
                      const float* __restrict__ conv_w,
                      const float* __restrict__ gamma,
                      const float* __restrict__ beta,
                      const float* __restrict__ rmean,
                      const float* __restrict__ rvar,
                      float* __restrict__ out)
{
    __shared__ float    s_x[3*NB][Dd][Cc];        // halo pixels, channel-fastest (18 KB)
    __shared__ _Float16 s_cost[TW][Dd+2][ICp];    // depth/channel zero-padded (25.5 KB)
    __shared__ _Float16 s_w[NCH][2][16][16];      // B fragments [chunk][g][oc][h] (9 KB)
    __shared__ float    s_dp[TW][Dd];             // center depth column (0.5 KB)

    const int h  = blockIdx.y;
    const int w0 = blockIdx.x * TW;
    const int t  = threadIdx.x;

    const float fx = intri[0], cx = intri[2];
    const float fy = intri[4], cy = intri[5];

    __builtin_prefetch(conv_w, 0, 3);   // global_prefetch_b8

    // ---------------- Phase 0: zero cost, stage x halo, dp, weight frags ----------------
    {
        v8h z8 = {};
        for (int idx = t; idx < TW*(Dd+2)*ICp/8; idx += 256)
            ((v8h*)s_cost)[idx] = z8;
    }
    for (int idx = t; idx < 3*NB*Cc*Dd; idx += 256) {
        int np = idx % (3*NB);          // halo pixel fastest -> lanes walk w
        int cz = idx / (3*NB);
        int c  = cz >> 5, z = cz & 31;
        int row = np / NB, col = np % NB;
        int hh  = iclamp(h  - 1 + row, 0, Hh - 1);   // edge padding
        int wg  = iclamp(w0 - 1 + col, 0, Ww - 1);
        s_x[np][z][c] = x[((c*Dd + z)*Hh + hh)*Ww + wg];
    }
    for (int idx = t; idx < TW*Dd; idx += 256) {
        int p = idx >> 5, z = idx & 31;
        s_dp[p][z] = dp[z*HW + h*Ww + (w0 + p)];
    }
    // weight fragments: chunk ch covers kd = ch/3, channels (ch%3)*32 .. +31
    for (int idx = t; idx < NCH*2*16*16; idx += 256) {
        int hh   = idx & 15;
        int rest = idx >> 4;
        int oc2  = rest & 15; rest >>= 4;
        int gg   = rest & 1;
        int ch   = rest >> 1;
        int r    = (hh & 7) + 8*gg + 16*(hh >> 3);  // B row within chunk
        int icp  = (ch % 3)*32 + r;         // kk-major padded channel
        int kd   = ch / 3;
        float wv = 0.f;
        if (icp < ICv && oc2 < OCn) {
            int kk = icp >> 3, c = icp & 7;
            wv = conv_w[(oc2*ICv + (c*K2 + kk))*3 + kd];
        }
        s_w[ch][gg][oc2][hh] = (_Float16)wv;
    }
    __syncthreads();

    // ---------------- Phase 1: build cost[dd][ic'] (f16), one b128 store/item ----------------
    const int NITEM = TW * K2 * Dd;
    for (int it = t; it < NITEM; it += 256) {
        int p  = it / (K2 * Dd);
        int r  = it % (K2 * Dd);
        int kk = r / Dd;
        int dd = r % Dd;
        int w  = w0 + p;
        int dy = kk / 3 - 1, dx = kk % 3 - 1;
        int hn = iclamp(h + dy, 0, Hh - 1);
        int wn = iclamp(w + dx, 0, Ww - 1);

        // planar weight ww = (n_nbr . pos_center) / (n_nbr . pos_nbr)
        float nx = normal[0*HW + hn*Ww + wn];
        float ny = normal[1*HW + hn*Ww + wn];
        float nz = normal[2*HW + hn*Ww + wn];
        float pxc = ((float)w  - cx) / fx, pyc = ((float)h  - cy) / fy;
        float pxn = ((float)wn - cx) / fx, pyn = ((float)hn - cy) / fy;
        float num = nx*pxc + ny*pyc + nz;
        float den = nx*pxn + ny*pyn + nz;
        if (fabsf(den) < 1e-8f) den = (den >= 0.f) ? 1e-8f : -1e-8f;
        float wwp = num / den;
        if (!__builtin_isfinite(wwp)) wwp = 1.0f;

        // z index (x/y grid coords are exact ints -> pure z-lerp at neighbor)
        float dp0 = dp[0*HW + hn*Ww + wn];
        float dp1 = dp[1*HW + hn*Ww + wn];
        float zi  = (wwp * s_dp[p][dd] - dp0) / (dp1 - dp0);
        float z   = fminf(fmaxf(zi, -1.f), (float)Dd);
        float z0f = floorf(z);
        float wz  = z - z0f;
        int   iz0 = (int)z0f;
        float w0z = (iz0     >= 0 && iz0     < Dd) ? (1.f - wz) : 0.f;
        float w1z = (iz0 + 1 >= 0 && iz0 + 1 < Dd) ? wz        : 0.f;
        int   iz0c = iclamp(iz0,     0, Dd - 1);
        int   iz1c = iclamp(iz0 + 1, 0, Dd - 1);

        int np = (dy + 1)*NB + (p + dx + 1);
        v8f xa = *(const v8f*)&s_x[np][iz0c][0];   // 8 channels, 2x ds_load_b128
        v8f xb = *(const v8f*)&s_x[np][iz1c][0];
        v8f vv = xa * w0z + xb * w1z;
        v8h pk;
#pragma unroll
        for (int c = 0; c < Cc; ++c) pk[c] = (_Float16)vv[c];
        *(v8h*)&s_cost[p][dd + 1][kk*8] = pk;      // one ds_store_b128
    }
    __syncthreads();

    // ---------------- Phase 2: WMMA depth-conv + BN + ReLU ----------------
    const int wave = t >> 5;
    const int lane = t & 31;
    const int pix  = wave >> 1;          // 4 pixels, 2 waves each
    const int d0   = (wave & 1) * 16;    // 16 depth rows per wave
    const int g    = lane >> 4;          // lane half-group
    const int oc   = lane & 15;          // B/D column
    const int m    = lane & 15;          // A/D row (depth position)

    // B fragments from LDS: two contiguous b128 loads each
    v16h bfr[NCH];
#pragma unroll
    for (int ch = 0; ch < NCH; ++ch) {
        const v8h* pb = (const v8h*)&s_w[ch][g][oc][0];
        v8h blo = pb[0], bhi = pb[1];
        bfr[ch] = __builtin_shufflevector(blo, bhi,
                    0,1,2,3,4,5,6,7,8,9,10,11,12,13,14,15);
    }

    v8f acc = {};
#pragma unroll
    for (int ch = 0; ch < NCH; ++ch) {
        const int kd  = ch / 3;
        const int icb = (ch % 3) * 32;
        // A row m, K halves: g=0 -> {0-7,16-23}, g=1 -> {8-15,24-31} (contiguous runs)
        const _Float16* row = &s_cost[pix][d0 + m + kd][0];
        v8h alo = *(const v8h*)&row[icb + 8*g];
        v8h ahi = *(const v8h*)&row[icb + 16 + 8*g];
        v16h a  = __builtin_shufflevector(alo, ahi,
                    0,1,2,3,4,5,6,7,8,9,10,11,12,13,14,15);
        acc = __builtin_amdgcn_wmma_f32_16x16x32_f16(
                  false, a, false, bfr[ch], (short)0, acc, false, false);
    }

    // D f32 layout: VGPR v, row M = v + 8*g, col N = oc
    if (oc < OCn) {
        float inv = rsqrtf(rvar[oc] + 1e-5f);
        float sc  = gamma[oc] * inv;
        float sh  = beta[oc] - rmean[oc] * sc;
        int   w   = w0 + pix;
#pragma unroll
        for (int v = 0; v < 8; ++v) {
            int d = d0 + v + 8*g;
            float yv = acc[v] * sc + sh;
            out[((oc*Dd + d)*Hh + h)*Ww + w] = fmaxf(yv, 0.f);
        }
    }
}

extern "C" void kernel_launch(void* const* d_in, const int* in_sizes, int n_in,
                              void* d_out, int out_size, void* d_ws, size_t ws_size,
                              hipStream_t stream) {
    const float* x      = (const float*)d_in[0];
    const float* dp     = (const float*)d_in[1];
    const float* normal = (const float*)d_in[2];
    const float* intri  = (const float*)d_in[3];
    const float* conv_w = (const float*)d_in[4];
    const float* gamma  = (const float*)d_in[5];
    const float* beta   = (const float*)d_in[6];
    const float* rmean  = (const float*)d_in[7];
    const float* rvar   = (const float*)d_in[8];
    float* out = (float*)d_out;

    dim3 grid(Ww / TW, Hh, 1);
    gca_fused_kernel<<<grid, 256, 0, stream>>>(x, dp, normal, intri, conv_w,
                                               gamma, beta, rmean, rvar, out);
}
